// GRU_35991825940739
// MI455X (gfx1250) — compile-verified, matrix operand order checked by reading, not run on hardware
//
#include <hip/hip_runtime.h>
#include <hip/hip_bf16.h>

// GRU, T=1024, B=256, I=H=O=256. Persistent fused bf16-WMMA implementation.
//
//   kernel 1 (pack_weights): fp32 -> bf16 WMMA B-fragment layout in d_ws
//       mats: 0=WxR 1=WhR 2=WxZ 3=WhZ 4=WxH 5=WhH 6=Wout; 8 ktiles x 16 ntiles;
//       fragment = 512 bf16 (1 KB), lane-major (lane*16 bf16 contiguous).
//   kernel 2 (gru_persistent): 16 WGs x 512 threads (16 waves); WG owns a
//       16-row batch slab for all T. Operand tiles live in bf16 LDS so an
//       A-fragment is 2x ds_load_b128; weights stream as global_load_b128
//       from the 192 MB L2 each step (per-iteration opaque *offset* defeats
//       LICM hoist/spill while preserving global address-space provenance).
//   Elementwise gate math uses hardware TRANS ops (v_exp_f32 / v_rcp_f32)
//   which co-execute with the XDL WMMA pipeline.

typedef __attribute__((ext_vector_type(16))) __bf16 v16bf;
typedef __attribute__((ext_vector_type(8)))  __bf16 v8bf;
typedef __attribute__((ext_vector_type(4)))  __bf16 v4bf;
typedef __attribute__((ext_vector_type(8)))  float  v8f;

#define T_STEPS 1024
#define BATCH   256
#define HID     256
#define MTILE   16
#define BST     280              // bf16 LDS row stride (560 B: 16B-aligned, bank-rotated)
#define HST     260              // fp32 h-master row stride (1040 B: 16B-aligned)
#define FRAG_ELEMS 512           // bf16 per 32x16 B fragment
#define MAT_ELEMS  (8 * 16 * FRAG_ELEMS)

// K offset (even) of pair p within a 32-wide K tile (CDNA5 16-bit A/B layout:
// lanes 0-15 hold K {0..7,16..23}; lanes 16-31 hold K {8..15,24..31}).
__device__ __forceinline__ int kOff(int p) {
    return (p < 4) ? (2 * p) : (16 + 2 * (p - 4));
}

// fast elementwise ops on hardware TRANS units
__device__ __forceinline__ float fast_sigmoid(float x) {
    return __builtin_amdgcn_rcpf(1.0f + __expf(-x));          // v_exp + v_rcp
}
__device__ __forceinline__ float fast_tanh(float x) {
    const float a = __builtin_fabsf(x);
    const float e = __expf(2.0f * a);                         // inf for large a -> t=1
    const float t = 1.0f - 2.0f * __builtin_amdgcn_rcpf(e + 1.0f);
    return __builtin_copysignf(t, x);
}

// ---------------- weight packing: fp32 [K,N] row-major -> bf16 fragments ----
__global__ __launch_bounds__(32)
void pack_weights(const float* __restrict__ wxr, const float* __restrict__ whr,
                  const float* __restrict__ wxz, const float* __restrict__ whz,
                  const float* __restrict__ wxh, const float* __restrict__ whh,
                  const float* __restrict__ wout, __bf16* __restrict__ wsW) {
    const float* mats[7] = {wxr, whr, wxz, whz, wxh, whh, wout};
    const int f    = blockIdx.x;        // 0..895
    const int mat  = f >> 7;
    const int rem  = f & 127;
    const int kt   = rem >> 4;
    const int nt   = rem & 15;
    const int lane = threadIdx.x;       // 0..31 (wave32)

    const float* W = mats[mat];
    const int n     = nt * 16 + (lane & 15);
    const int kbase = kt * 32 + ((lane >> 4) ? 8 : 0);
    __bf16* dst = wsW + (size_t)mat * MAT_ELEMS + (size_t)(kt * 16 + nt) * FRAG_ELEMS
                      + lane * 16;
#pragma unroll
    for (int p = 0; p < 8; ++p) {
        const int k = kbase + kOff(p);
        dst[2 * p]     = (__bf16)W[(size_t)k       * HID + n];
        dst[2 * p + 1] = (__bf16)W[(size_t)(k + 1) * HID + n];
    }
}

// ---------------- fragment helpers --------------------------------------
// A fragment from a bf16 LDS tile: two 16-byte K-chunks per lane.
__device__ __forceinline__ v16bf packA(const __bf16* S, int kt, int lane) {
    const int M   = lane & 15;
    const int off = kt * 32 + ((lane >> 4) ? 8 : 0);
    const __bf16* p = S + M * BST + off;
    const v8bf lo = *(const v8bf*)p;          // ds_load_b128 (K off+0..7)
    const v8bf hi = *(const v8bf*)(p + 16);   // ds_load_b128 (K off+16..23)
    return __builtin_shufflevector(lo, hi, 0, 1, 2, 3, 4, 5, 6, 7,
                                           8, 9, 10, 11, 12, 13, 14, 15);
}

__device__ __forceinline__ v16bf loadB(const __bf16* __restrict__ wsW, size_t toff,
                                       int mat, int kt, int nt, int lane) {
    const __bf16* p = wsW + toff + (size_t)mat * MAT_ELEMS
                          + (size_t)(kt * 16 + nt) * FRAG_ELEMS + lane * 16;
    return *(const v16bf*)p;                  // 2x global_load_b128 (L2-hot)
}

#define WMMA_BF16(A, B, C) \
    __builtin_amdgcn_wmma_f32_16x16x32_bf16(false, (A), false, (B), (short)0, (C), false, false)

// stage a fp32 global tile [16,256] into a bf16 LDS tile (coalesced b128 loads)
__device__ __forceinline__ void stageX(const float* __restrict__ g,
                                       __bf16* xB, int tid) {
#pragma unroll
    for (int it = 0; it < 2; ++it) {
        const int i4  = tid + it * 512;       // float4 index, 0..1023
        const int row = i4 >> 6;
        const int c4  = i4 & 63;
        const float4 v = *(const float4*)(g + (size_t)row * HID + c4 * 4);
        const v4bf b = {(__bf16)v.x, (__bf16)v.y, (__bf16)v.z, (__bf16)v.w};
        *(v4bf*)(xB + row * BST + c4 * 4) = b;
    }
}

// ---------------- persistent fused GRU ----------------------------------
__global__ __launch_bounds__(512)
void gru_persistent(const float* __restrict__ x, const float* __restrict__ h0,
                    const float* __restrict__ b_r, const float* __restrict__ b_z,
                    const float* __restrict__ b_h, const float* __restrict__ b_out,
                    const __bf16* __restrict__ wsW, float* __restrict__ out) {
    __shared__ __bf16 xB [MTILE * BST];   // x_t tile, bf16
    __shared__ __bf16 hB [MTILE * BST];   // h tile, bf16 (GEMM operand copy)
    __shared__ __bf16 rhB[MTILE * BST];   // r*h tile, bf16
    __shared__ __bf16 cB [MTILE * BST];   // cand tile, bf16
    __shared__ float  hF [MTILE * HST];   // fp32 master h (exact update)

    const int tid  = threadIdx.x;
    const int lane = tid & 31;
    const int nt   = tid >> 5;                 // wave id == N tile (0..15)
    const int m0   = blockIdx.x * MTILE;       // batch slab base row

    // ---- stage h0 (fp32 master + bf16 copy) and x(t=0) ----
#pragma unroll
    for (int it = 0; it < 2; ++it) {
        const int i4  = tid + it * 512;
        const int row = i4 >> 6;
        const int c4  = i4 & 63;
        const float4 v = *(const float4*)(h0 + (size_t)(m0 + row) * HID + c4 * 4);
        *(float4*)(hF + row * HST + c4 * 4) = v;
        const v4bf b = {(__bf16)v.x, (__bf16)v.y, (__bf16)v.z, (__bf16)v.w};
        *(v4bf*)(hB + row * BST + c4 * 4) = b;
    }
    stageX(x + (size_t)m0 * HID, xB, tid);

    const int   col    = nt * 16 + (lane & 15);   // this lane's output column
    const int   rowsel = (lane >> 4) ? 8 : 0;     // C/D layout: +8 rows for hi lanes
    const float brv = b_r[col], bzv = b_z[col], bhv = b_h[col], bov = b_out[col];

    __syncthreads();

    for (int t = 0; t < T_STEPS; ++t) {
        // Opaque per-iteration *offset* (always 0): defeats LICM hoist/spill of
        // the 56 B-fragments while keeping the base pointer's global-addrspace
        // provenance -> global_load_b128 (LOADcnt only, decoupled from DScnt).
        size_t toff = 0;
        asm volatile("" : "+s"(toff));

        // ---- gates r, z ------------------------------------------------
        v8f accR = {}; v8f accZ = {};
#pragma unroll
        for (int kt = 0; kt < 8; ++kt) {
            const v16bf ax = packA(xB, kt, lane);
            accR = WMMA_BF16(ax, loadB(wsW, toff, 0, kt, nt, lane), accR);
            accZ = WMMA_BF16(ax, loadB(wsW, toff, 2, kt, nt, lane), accZ);
            const v16bf ah = packA(hB, kt, lane);
            accR = WMMA_BF16(ah, loadB(wsW, toff, 1, kt, nt, lane), accR);
            accZ = WMMA_BF16(ah, loadB(wsW, toff, 3, kt, nt, lane), accZ);
        }
        float zv[8];
#pragma unroll
        for (int i = 0; i < 8; ++i) {
            const float r = fast_sigmoid(accR[i] + brv);
            zv[i]         = fast_sigmoid(accZ[i] + bzv);
            const int M = i + rowsel;
            rhB[M * BST + col] = (__bf16)(r * hF[M * HST + col]);
        }
        __syncthreads();   // B1: rhB ready; all gate reads of hB done

        // ---- candidate -------------------------------------------------
        v8f accC = {};
#pragma unroll
        for (int kt = 0; kt < 8; ++kt) {
            accC = WMMA_BF16(packA(xB,  kt, lane), loadB(wsW, toff, 4, kt, nt, lane), accC);
            accC = WMMA_BF16(packA(rhB, kt, lane), loadB(wsW, toff, 5, kt, nt, lane), accC);
        }
#pragma unroll
        for (int i = 0; i < 8; ++i) {
            const float cand = fast_tanh(accC[i] + bhv);
            const int M = i + rowsel;
            cB[M * BST + col] = (__bf16)cand;
            const float hNew = hF[M * HST + col] * zv[i] + (1.0f - zv[i]) * cand;
            hF[M * HST + col] = hNew;            // exact fp32 state
            hB[M * BST + col] = (__bf16)hNew;    // GEMM operand copy
        }
        __syncthreads();   // B2: cB/hB ready; xB free for next-t overwrite

        // ---- output GEMM: cand @ Wout + b_out --------------------------
        v8f accO = {};
#pragma unroll
        for (int kt = 0; kt < 8; ++kt)
            accO = WMMA_BF16(packA(cB, kt, lane), loadB(wsW, toff, 6, kt, nt, lane), accO);

        float* orow = out + ((size_t)t * BATCH + m0) * HID;
#pragma unroll
        for (int i = 0; i < 8; ++i) {
            const int M = i + rowsel;
            orow[(size_t)M * HID + col] = accO[i] + bov;
        }

        // ---- stage x(t+1), prefetch x(t+2) -----------------------------
        if (t + 1 < T_STEPS) {
            stageX(x + ((size_t)(t + 1) * BATCH + m0) * HID, xB, tid);
            if (t + 2 < T_STEPS)
                __builtin_prefetch(x + ((size_t)(t + 2) * BATCH + m0) * HID + tid * 8, 0, 1);
        }
        __syncthreads();   // B3: xB ready, step fully retired
    }

    // h_final appended after the T*B*O outputs
    float* hf = out + (size_t)T_STEPS * BATCH * HID;
#pragma unroll
    for (int it = 0; it < 8; ++it) {
        const int idx = tid + it * 512;       // 0..4095
        const int row = idx >> 8;
        const int c   = idx & 255;
        hf[(size_t)(m0 + row) * HID + c] = hF[row * HST + c];
    }
}

// ---------------- host entry ---------------------------------------------
extern "C" void kernel_launch(void* const* d_in, const int* in_sizes, int n_in,
                              void* d_out, int out_size, void* d_ws, size_t ws_size,
                              hipStream_t stream) {
    const float* x     = (const float*)d_in[0];
    const float* h0    = (const float*)d_in[1];
    const float* w_xr  = (const float*)d_in[2];
    const float* w_hr  = (const float*)d_in[3];
    const float* b_r   = (const float*)d_in[4];
    const float* w_xz  = (const float*)d_in[5];
    const float* w_hz  = (const float*)d_in[6];
    const float* b_z   = (const float*)d_in[7];
    const float* w_xh  = (const float*)d_in[8];
    const float* w_hh  = (const float*)d_in[9];
    const float* b_h   = (const float*)d_in[10];
    const float* w_out = (const float*)d_in[11];
    const float* b_out = (const float*)d_in[12];

    __bf16* wsW = (__bf16*)d_ws;          // 7 * 65536 bf16 = 896 KB scratch
    float*  out = (float*)d_out;

    pack_weights<<<7 * 8 * 16, 32, 0, stream>>>(w_xr, w_hr, w_xz, w_hz,
                                                w_xh, w_hh, w_out, wsW);
    gru_persistent<<<BATCH / MTILE, 512, 0, stream>>>(x, h0, b_r, b_z, b_h, b_out,
                                                      wsW, out);
}
